// SwinTransformerBlock_67662914781970
// MI455X (gfx1250) — matrix operand order repaired and checked by reading, not compiled, not run on hardware
//
#include <hip/hip_runtime.h>
#include <hip/hip_bf16.h>

// ---------------------------------------------------------------------------
// Swin3D block for MI455X (gfx1250, wave32, WMMA).
// All GEMMs use v_wmma_f32_16x16x32_f16 (f16 A/B, f32 accumulate).
// ---------------------------------------------------------------------------

typedef __attribute__((ext_vector_type(16))) _Float16 v16h;
typedef __attribute__((ext_vector_type(8)))  float    v8f;

union Frag16 { v16h v; uint4 q[2]; };

__device__ __forceinline__ v8f wmma_f16(v16h a, v16h b, v8f c) {
    return __builtin_amdgcn_wmma_f32_16x16x32_f16(false, a, false, b, (short)0, c, false, false);
}

// A-fragment (16x32 f16) from row-major matrix. Layout:
// lane<16: K = 0..15; lane>=16: K = 16..31.
__device__ __forceinline__ v16h load_afrag(const _Float16* base, int row, int stride,
                                           int k0, int lane) {
    const _Float16* p = base + row * stride + k0 + ((lane >> 4) << 4);
    Frag16 f;
    f.q[0] = *(const uint4*)(p);
    f.q[1] = *(const uint4*)(p + 8);
    return f.v;
}

// B-fragment (32x16) from a row-major matrix whose ROWS are the B columns
// (i.e. B^T stored row-major). lane<16 holds K=0..15, lane>=16 K=16..31.
__device__ __forceinline__ v16h load_bfragT(const _Float16* base, int col, int stride,
                                            int k0, int lane) {
    const _Float16* p = base + col * stride + k0 + ((lane >> 4) << 4);
    Frag16 f;
    f.q[0] = *(const uint4*)(p);
    f.q[1] = *(const uint4*)(p + 8);
    return f.v;
}

// B-fragment from pre-packed weights: layout [kt][nt][lane][8 dwords] (32B/lane).
__device__ __forceinline__ v16h load_bfrag_packed(const unsigned int* pw, int kt, int nt,
                                                  int NT, int lane) {
    const unsigned int* p = pw + ((((kt * NT + nt) << 5) + lane) << 3);
    Frag16 f;
    f.q[0] = *(const uint4*)(p);
    f.q[1] = *(const uint4*)(p + 4);
    return f.v;
}

// One wave (32 lanes) layernorms one 192-channel token into LDS (f16).
__device__ __forceinline__ void layernorm_row(const float* __restrict__ src,
                                              const float* __restrict__ g,
                                              const float* __restrict__ bta,
                                              _Float16* dst, int lane) {
    float v[6];
    float s = 0.f, sq = 0.f;
#pragma unroll
    for (int i = 0; i < 6; ++i) { v[i] = src[lane + (i << 5)]; s += v[i]; sq += v[i] * v[i]; }
#pragma unroll
    for (int m = 16; m; m >>= 1) { s += __shfl_xor(s, m, 32); sq += __shfl_xor(sq, m, 32); }
    float mean = s * (1.f / 192.f);
    float var  = sq * (1.f / 192.f) - mean * mean;
    float inv  = rsqrtf(var + 1e-5f);
#pragma unroll
    for (int i = 0; i < 6; ++i) {
        int c = lane + (i << 5);
        dst[c] = (_Float16)((v[i] - mean) * inv * g[c] + bta[c]);
    }
}

// ---------------------------------------------------------------------------
// Weight packing: W (KxN f32, row-major) -> fragment order f16 pairs.
// dword id = ((kt*NT + nt)*32 + lane)*8 + v holds W[k][n],W[k+1][n] with
// k = kt*32 + (lane/16)*16 + 2v, n = nt*16 + (lane%16).
// ---------------------------------------------------------------------------
__global__ void pack_kernel(const float* __restrict__ W, unsigned int* __restrict__ P,
                            int K, int N) {
    int id = blockIdx.x * 256 + threadIdx.x;
    int total = (K * N) >> 1;
    if (id >= total) return;
    int v = id & 7;
    int l = (id >> 3) & 31;
    int tile = id >> 8;
    int NT = N >> 4;
    int nt = tile % NT, kt = tile / NT;
    int k = kt * 32 + ((l >> 4) << 4) + (v << 1);
    int n = nt * 16 + (l & 15);
    union { _Float16 h[2]; unsigned int u; } cv;
    cv.h[0] = (_Float16)W[k * N + n];
    cv.h[1] = (_Float16)W[(k + 1) * N + n];
    P[id] = cv.u;
}

// bias[h][l][m] = table[rel_pos_index[l*64+m]][h]   (6*64*64 floats)
__global__ void bias_kernel(const float* __restrict__ table, const int* __restrict__ rpi,
                            float* __restrict__ biasbuf) {
    int id = blockIdx.x * 256 + threadIdx.x;   // 24576 total
    int rem = id & 4095;
    int h = id >> 12;
    biasbuf[id] = table[rpi[rem] * 6 + h];
}

// ---------------------------------------------------------------------------
// Attention kernel: one workgroup (8 waves) per window.
// LN1(+roll,+partition) -> per-head QKV -> S=QK^T+bias+mask -> softmax -> PV.
// ---------------------------------------------------------------------------
__global__ __launch_bounds__(256) void attn_kernel(
    const float* __restrict__ x, const float* __restrict__ g1, const float* __restrict__ b1,
    const unsigned int* __restrict__ pWqkv, const float* __restrict__ bqkv,
    const float* __restrict__ biasbuf, const unsigned char* __restrict__ mask,
    _Float16* __restrict__ gO)
{
    __shared__ _Float16 sA[64 * 192];   // 24 KB LN'd window tokens
    __shared__ _Float16 sQ[64 * 32];    // 4 KB (q * scale, this head)
    __shared__ _Float16 sK[64 * 32];    // 4 KB
    __shared__ _Float16 sVT[32 * 64];   // 4 KB (v transposed: [ch][token])
    __shared__ float    sS[64 * 64];    // 16 KB scores; f16 P aliased in-place

    const int tid = threadIdx.x;
    const int lane = tid & 31;
    const int wave = tid >> 5;
    const int b = blockIdx.x;
    const int img = b >> 9;
    const int w = b & 511;
    const int w0 = w >> 6, w1 = (w >> 3) & 7, w2 = w & 7;

    // Phase 0: LN1 with roll(+2,+2,+2) gather, window partition into sA.
#pragma unroll
    for (int i = 0; i < 8; ++i) {
        int r = wave * 8 + i;
        int i0 = r >> 4, i1 = (r >> 2) & 3, i2 = r & 3;
        int sd = ((w0 << 2) + i0 + 30) & 31;
        int sh = ((w1 << 2) + i1 + 30) & 31;
        int sw = ((w2 << 2) + i2 + 30) & 31;
        const float* src = x + ((((size_t)img * 32 + sd) * 32 + sh) * 32 + sw) * 192;
        layernorm_row(src, g1, b1, sA + r * 192, lane);
    }
    __syncthreads();

    for (int hd = 0; hd < 6; ++hd) {
        // Phase 1: QKV for this head: 24 tiles (4 mt x {q,k,v} x 2 col halves).
#pragma unroll
        for (int i = 0; i < 3; ++i) {
            int t = wave + (i << 3);
            int mt = t / 6, rem = t % 6;
            int which = rem >> 1, nh = rem & 1;
            int ntg = which * 12 + hd * 2 + nh;    // col tile in 576 (36 tiles)
            v8f acc = {};
#pragma unroll
            for (int kt = 0; kt < 6; ++kt) {
                v16h a = load_afrag(sA, mt * 16 + (lane & 15), 192, kt * 32, lane);
                v16h bf = load_bfrag_packed(pWqkv, kt, ntg, 36, lane);
                acc = wmma_f16(a, bf, acc);
            }
            float bb = bqkv[ntg * 16 + (lane & 15)];
            int colh = nh * 16 + (lane & 15);
#pragma unroll
            for (int v = 0; v < 8; ++v) {
                int row = mt * 16 + ((lane >> 4) << 3) + v;
                float val = acc[v] + bb;
                if (which == 0)      sQ[row * 32 + colh] = (_Float16)(val * 0.17677669529663687f);
                else if (which == 1) sK[row * 32 + colh] = (_Float16)val;
                else                 sVT[colh * 64 + row] = (_Float16)val;
            }
        }
        __syncthreads();

        // Phase 2: S = q @ k^T  (K = CH = 32 -> one WMMA per 16x16 tile).
#pragma unroll
        for (int i = 0; i < 2; ++i) {
            int t = wave + (i << 3);
            int mt = t >> 2, nt = t & 3;
            v16h a  = load_afrag(sQ, mt * 16 + (lane & 15), 32, 0, lane);
            v16h bf = load_bfragT(sK, nt * 16 + (lane & 15), 32, 0, lane);
            v8f acc = {};
            acc = wmma_f16(a, bf, acc);
#pragma unroll
            for (int v = 0; v < 8; ++v) {
                int m = mt * 16 + ((lane >> 4) << 3) + v;
                int n = nt * 16 + (lane & 15);
                float val = acc[v] + biasbuf[(hd * 64 + m) * 64 + n];
                if (mask[((size_t)w * 64 + m) * 64 + n]) val = -1e30f;
                sS[m * 64 + n] = val;
            }
        }
        __syncthreads();

        // Phase 3: row softmax, 4 lanes per row, P written f16 in-place over sS.
        {
            int row = tid >> 2, part = tid & 3;
            float e[16];
            float mx = -1e30f;
#pragma unroll
            for (int j = 0; j < 16; ++j) { e[j] = sS[row * 64 + part * 16 + j]; mx = fmaxf(mx, e[j]); }
            mx = fmaxf(mx, __shfl_xor(mx, 1, 32));
            mx = fmaxf(mx, __shfl_xor(mx, 2, 32));
            float s = 0.f;
#pragma unroll
            for (int j = 0; j < 16; ++j) { e[j] = __expf(e[j] - mx); s += e[j]; }
            s += __shfl_xor(s, 1, 32);
            s += __shfl_xor(s, 2, 32);
            float inv = 1.f / s;
            __syncthreads();   // all reads of sS done before f16 overwrite
            _Float16* prow = (_Float16*)(sS + row * 64);   // P stride = 128 halves
#pragma unroll
            for (int j = 0; j < 16; ++j) prow[part * 16 + j] = (_Float16)(e[j] * inv);
        }
        __syncthreads();

        // Phase 4: O_head = P @ V : 8 tiles, one per wave. K = 64 -> 2 WMMAs.
        {
            int mt = wave >> 1, nt = wave & 1;
            const _Float16* P = (const _Float16*)sS;
            v8f acc = {};
#pragma unroll
            for (int kt = 0; kt < 2; ++kt) {
                v16h a  = load_afrag(P, mt * 16 + (lane & 15), 128, kt * 32, lane);
                v16h bf = load_bfragT(sVT, nt * 16 + (lane & 15), 64, kt * 32, lane);
                acc = wmma_f16(a, bf, acc);
            }
#pragma unroll
            for (int v = 0; v < 8; ++v) {
                int row = mt * 16 + ((lane >> 4) << 3) + v;
                int col = hd * 32 + nt * 16 + (lane & 15);
                gO[((size_t)b * 64 + row) * 192 + col] = (_Float16)acc[v];
            }
        }
        __syncthreads();
    }
}

// ---------------------------------------------------------------------------
// Projection: o @ wproj + bproj, window-reverse + un-roll, residual -> x2.
// ---------------------------------------------------------------------------
__global__ __launch_bounds__(256) void proj_kernel(
    const float* __restrict__ x, const _Float16* __restrict__ gO,
    const unsigned int* __restrict__ pWproj, const float* __restrict__ bproj,
    float* __restrict__ x2)
{
    __shared__ _Float16 sO[64 * 192];  // 24 KB
    const int tid = threadIdx.x;
    const int lane = tid & 31;
    const int wave = tid >> 5;
    const int b = blockIdx.x;
    const int img = b >> 9;
    const int w = b & 511;
    const int w0 = w >> 6, w1 = (w >> 3) & 7, w2 = w & 7;

    const uint4* src = (const uint4*)(gO + (size_t)b * 64 * 192);
    uint4* dst = (uint4*)sO;
#pragma unroll
    for (int i = tid; i < 1536; i += 256) dst[i] = src[i];
    __syncthreads();

#pragma unroll
    for (int i = 0; i < 6; ++i) {
        int t = wave + (i << 3);
        int mt = t / 12, nt = t % 12;
        v8f acc = {};
#pragma unroll
        for (int kt = 0; kt < 6; ++kt) {
            v16h a  = load_afrag(sO, mt * 16 + (lane & 15), 192, kt * 32, lane);
            v16h bf = load_bfrag_packed(pWproj, kt, nt, 12, lane);
            acc = wmma_f16(a, bf, acc);
        }
        int c = nt * 16 + (lane & 15);
        float bb = bproj[c];
#pragma unroll
        for (int v = 0; v < 8; ++v) {
            int r = mt * 16 + ((lane >> 4) << 3) + v;
            int i0 = r >> 4, i1 = (r >> 2) & 3, i2 = r & 3;
            int dd = ((w0 << 2) + i0 + 30) & 31;   // un-roll by -2
            int hh = ((w1 << 2) + i1 + 30) & 31;
            int ww = ((w2 << 2) + i2 + 30) & 31;
            size_t idx = ((((size_t)img * 32 + dd) * 32 + hh) * 32 + ww) * 192 + c;
            x2[idx] = x[idx] + acc[v] + bb;
        }
    }
}

// ---------------------------------------------------------------------------
// MLP1: LN2(x2) @ wm1 + bm1 -> erf-GELU -> gM (f16).  64 rows per workgroup.
// ---------------------------------------------------------------------------
__global__ __launch_bounds__(256) void mlp1_kernel(
    const float* __restrict__ x2, const float* __restrict__ g2, const float* __restrict__ b2,
    const unsigned int* __restrict__ pWm1, const float* __restrict__ bm1,
    _Float16* __restrict__ gM)
{
    __shared__ _Float16 sH[64 * 192];  // 24 KB
    const int tid = threadIdx.x, lane = tid & 31, wave = tid >> 5;
    const size_t row0 = (size_t)blockIdx.x * 64;

#pragma unroll
    for (int i = 0; i < 8; ++i) {
        int r = wave * 8 + i;
        layernorm_row(x2 + (row0 + r) * 192, g2, b2, sH + r * 192, lane);
    }
    __syncthreads();

#pragma unroll 4
    for (int i = 0; i < 24; ++i) {
        int t = wave + (i << 3);
        int mt = t / 48, nt = t % 48;
        v8f acc = {};
#pragma unroll
        for (int kt = 0; kt < 6; ++kt) {
            v16h a  = load_afrag(sH, mt * 16 + (lane & 15), 192, kt * 32, lane);
            v16h bf = load_bfrag_packed(pWm1, kt, nt, 48, lane);
            acc = wmma_f16(a, bf, acc);
        }
        int c = nt * 16 + (lane & 15);
        float bb = bm1[c];
#pragma unroll
        for (int v = 0; v < 8; ++v) {
            int r = mt * 16 + ((lane >> 4) << 3) + v;
            float h = acc[v] + bb;
            float gel = 0.5f * h * (1.f + erff(h * 0.7071067811865476f));
            gM[(row0 + r) * 768 + c] = (_Float16)gel;
        }
    }
}

// ---------------------------------------------------------------------------
// MLP2: gM @ wm2 + bm2 + x2 -> out (f32). K=768 staged in 4 chunks of 192.
// Each wave holds 6 f32 accumulator tiles across the whole K loop.
// ---------------------------------------------------------------------------
__global__ __launch_bounds__(256) void mlp2_kernel(
    const _Float16* __restrict__ gM, const unsigned int* __restrict__ pWm2,
    const float* __restrict__ bm2, const float* __restrict__ x2,
    float* __restrict__ out)
{
    __shared__ _Float16 sM[64 * 192];  // 24 KB K-chunk staging
    const int tid = threadIdx.x, lane = tid & 31, wave = tid >> 5;
    const size_t row0 = (size_t)blockIdx.x * 64;

    v8f acc[6] = {};
    for (int kc = 0; kc < 4; ++kc) {
        __syncthreads();
        for (int i = tid; i < 1536; i += 256) {
            int r = i / 24, cq = i % 24;
            ((uint4*)sM)[i] = *(const uint4*)(gM + (row0 + r) * 768 + kc * 192 + cq * 8);
        }
        __syncthreads();
#pragma unroll
        for (int i = 0; i < 6; ++i) {
            int t = wave + (i << 3);
            int mt = t & 3;
#pragma unroll
            for (int ks = 0; ks < 6; ++ks) {
                v16h a  = load_afrag(sM, mt * 16 + (lane & 15), 192, ks * 32, lane);
                v16h bf = load_bfrag_packed(pWm2, kc * 6 + ks, t >> 2, 12, lane);
                acc[i] = wmma_f16(a, bf, acc[i]);
            }
        }
    }
#pragma unroll
    for (int i = 0; i < 6; ++i) {
        int t = wave + (i << 3);
        int mt = t & 3, nt = t >> 2;
        int c = nt * 16 + (lane & 15);
        float bb = bm2[c];
#pragma unroll
        for (int v = 0; v < 8; ++v) {
            size_t r = row0 + mt * 16 + ((lane >> 4) << 3) + v;
            out[r * 192 + c] = x2[r * 192 + c] + acc[i][v] + bb;
        }
    }
}

// ---------------------------------------------------------------------------
// Host launcher.
// ---------------------------------------------------------------------------
extern "C" void kernel_launch(void* const* d_in, const int* in_sizes, int n_in,
                              void* d_out, int out_size, void* d_ws, size_t ws_size,
                              hipStream_t stream) {
    (void)in_sizes; (void)n_in; (void)out_size; (void)ws_size;
    const float* x      = (const float*)d_in[0];
    const int*   rpi    = (const int*)d_in[1];
    const unsigned char* mask = (const unsigned char*)d_in[2];
    const float* table  = (const float*)d_in[3];
    const float* wqkv   = (const float*)d_in[4];
    const float* bqkv   = (const float*)d_in[5];
    const float* wproj  = (const float*)d_in[6];
    const float* bproj  = (const float*)d_in[7];
    const float* g1     = (const float*)d_in[8];
    const float* b1     = (const float*)d_in[9];
    const float* g2     = (const float*)d_in[10];
    const float* b2     = (const float*)d_in[11];
    const float* wm1    = (const float*)d_in[12];
    const float* bm1    = (const float*)d_in[13];
    const float* wm2    = (const float*)d_in[14];
    const float* bm2    = (const float*)d_in[15];

    // Workspace carve-up (bytes, 256-aligned).
    char* ws = (char*)d_ws;
    unsigned int* pWqkv  = (unsigned int*)(ws + 0);            //  55296 dw
    unsigned int* pWproj = (unsigned int*)(ws + 221184);       //  18432 dw
    unsigned int* pWm1   = (unsigned int*)(ws + 294912);       //  73728 dw
    unsigned int* pWm2   = (unsigned int*)(ws + 589824);       //  73728 dw
    float*        biasb  = (float*)      (ws + 884736);        //  24576 f32
    _Float16*     gO     = (_Float16*)   (ws + 983040);        //  1024*64*192 f16
    float*        x2     = (float*)      (ws + 26148864);      //  65536*192 f32
    _Float16*     gM     = (_Float16*)   (ws + 76480512);      //  65536*768 f16
    float*        outp   = (float*)d_out;

    pack_kernel<<<216, 256, 0, stream>>>(wqkv,  pWqkv, 192, 576);
    pack_kernel<<< 72, 256, 0, stream>>>(wproj, pWproj, 192, 192);
    pack_kernel<<<288, 256, 0, stream>>>(wm1,   pWm1, 192, 768);
    pack_kernel<<<288, 256, 0, stream>>>(wm2,   pWm2, 768, 192);
    bias_kernel<<< 96, 256, 0, stream>>>(table, rpi, biasb);

    attn_kernel<<<1024, 256, 0, stream>>>(x, g1, b1, pWqkv, bqkv, biasb, mask, gO);
    proj_kernel<<<1024, 256, 0, stream>>>(x, gO, pWproj, bproj, x2);
    mlp1_kernel<<<1024, 256, 0, stream>>>(x2, g2, b2, pWm1, bm1, gM);
    mlp2_kernel<<<1024, 256, 0, stream>>>(gM, pWm2, bm2, x2, outp);
}
